// AssociativeLIF_25262997636004
// MI455X (gfx1250) — matrix operand order repaired and checked by reading, not compiled, not run on hardware
//
#include <hip/hip_runtime.h>

// CDNA5 (gfx1250) — wave32, WMMA 16x16x32 f16 -> f32.
typedef __attribute__((ext_vector_type(16))) _Float16 v16h;
typedef __attribute__((ext_vector_type(8)))  float    v8f;

#define T_STEPS 512
#define N_NEUR  1024

// Workgroup barrier with LDS-only visibility: waits DScnt==0 (so our ds_store
// is visible to other waves) but does NOT drain global load/store counters the
// way __syncthreads() does.  This keeps the register double-buffered x loads
// and the spike stores in flight across the per-step barriers.
__device__ __forceinline__ void barrier_lds_only() {
  asm volatile("s_wait_dscnt 0x0\n\t"
               "s_barrier_signal -1\n\t"
               "s_barrier_wait -1" ::: "memory");
}

// Sum across the 4 adjacent lanes of a quad via DPP quad_perm (pure VALU; no
// LDS round-trip like ds_bpermute).  0xB1 = perm[1,0,3,2] (xor 1),
// 0x4E = perm[2,3,0,1] (xor 2).
__device__ __forceinline__ float quad_fold_add(float x) {
  int a = __builtin_amdgcn_update_dpp(0, __float_as_int(x), 0xB1, 0xF, 0xF, true);
  float s = x + __int_as_float(a);
  int b2 = __builtin_amdgcn_update_dpp(0, __float_as_int(s), 0x4E, 0xF, 0xF, true);
  return s + __int_as_float(b2);
}

// One workgroup per batch row (64 WGs -> 64 WGPs, max legal parallelism: the
// cascade couples all 1024 neurons of a row every step).  256 threads: thread
// (c = tid>>2, h = tid&3) owns neurons n = c + 64*(h+4*jj), jj = 0..3.
// Cascade ns[c] = gain[c] * sum_k cf[k]*sigmoid(W)[c,k] runs as a 16x64 @ 64x64
// WMMA (A rows M=1..15 pinned to zero at init).
__launch_bounds__(256, 1)
__global__ void lif_assoc_kernel(const float* __restrict__ x,
                                 const float* __restrict__ vth_raw,
                                 const float* __restrict__ bmem_raw,
                                 const float* __restrict__ bsyn_raw,
                                 const float* __restrict__ nw,
                                 const float* __restrict__ gain,
                                 float* __restrict__ out) {
  __shared__ unsigned int ldsA[2][32][8];   // f16-packed WMMA A frags (two K=32 halves)
  __shared__ float ldsNS[64];               // cascade result ns[cluster]

  const int tid = threadIdx.x;
  const int c   = tid >> 2;                 // cluster id == K index (0..63)
  const int h   = tid & 3;                  // j-subgroup within cluster
  const int b   = blockIdx.x;               // batch row

  // Zero the A-fragment image once; only M=0 slots are rewritten per step.
  ((unsigned int*)ldsA)[tid]       = 0u;
  ((unsigned int*)ldsA)[tid + 256] = 0u;

  // Scalar parameter transforms (match reference clipping).
  float bmem = 1.0f / (1.0f + __expf(-bmem_raw[0]));
  bmem = fminf(fmaxf(bmem, 0.80f), 0.98f);
  const float bsyn = 1.0f / (1.0f + __expf(-bsyn_raw[0]));
  const float omb  = 1.0f - bmem;

  // Per-neuron state in VGPRs for the whole T loop.
  float v[4], isyn[4], thv[4];
  int refrac[4];
#pragma unroll
  for (int jj = 0; jj < 4; ++jj) {
    v[jj] = 0.0f; isyn[jj] = 0.0f; refrac[jj] = 0;
    thv[jj] = fminf(fmaxf(vth_raw[c + 64 * (h + 4 * jj)], 0.05f), 0.50f);
  }

  // A-fragment slot for (M=0, K=c)  (ISA 7.12.2 16-bit A 16x32 layout:
  // lane = M + 16*K[3]; vgpr = (K&7)/2 + 4*K[4]; f16 half = K&1).
  const int kh    = c >> 5;
  const int kk    = c & 31;
  const int alane = ((kk >> 3) & 1) << 4;
  const int advr  = ((kk & 7) >> 1) + ((kk & 16) ? 4 : 0);
  _Float16* aslot = ((_Float16*)&ldsA[kh][alane][advr]) + (kk & 1);

  // WMMA wave setup; scalarized wave id keeps EXEC all-ones inside the branch.
  const int wq = __builtin_amdgcn_readfirstlane(tid >> 5);
  const int ll = tid & 31;
  v16h bf0 = {}, bf1 = {};
  if (wq < 4) {
    // B[k][cOut] = sigmoid(nw[cOut][k]) * gain[cOut]; tile cols [16*wq,16*wq+16).
    // 32x16 B layout: vgpr i<4, lanes 0-15: K=2i,2i+1 ; lanes 16-31: K=8+2i,9+2i; vgpr i>=4: K+8.
    const int Ng  = wq * 16 + (ll & 15);
    const float g = gain[Ng];
    const int kb  = (ll < 16) ? 0 : 8;
#pragma unroll
    for (int e = 0; e < 16; ++e) {
      bf0[e] = (_Float16)(g / (1.0f + __expf(-nw[Ng * 64 + (kb + e + (e >= 8 ? 8 : 0))])));
      bf1[e] = (_Float16)(g / (1.0f + __expf(-nw[Ng * 64 + (32 + kb + e + (e >= 8 ? 8 : 0))])));
    }
  }
  __syncthreads();   // full sync once: ldsA zero-init visible before the loop

  const float* xt = x   + (size_t)b * T_STEPS * N_NEUR + c + 64 * h;
  float*       ot = out + (size_t)b * T_STEPS * N_NEUR + c + 64 * h;

  // Double-buffer x in registers: step t+1's loads stay in flight across the
  // barrier/WMMA phase (barriers no longer drain LOADcnt).
  float xv[4];
#pragma unroll
  for (int jj = 0; jj < 4; ++jj) xv[jj] = xt[256 * jj];

  for (int t = 0; t < T_STEPS; ++t) {
    // Branchless clamp: last iteration harmlessly reloads the final row
    // (keeps a single loop body; no OOB access, no OOB prefetch).
    const float* xnp = (t < T_STEPS - 1) ? (xt + N_NEUR) : xt;
    const float* xpf = (t < T_STEPS - 2) ? (xt + 2 * N_NEUR) : xt;
    __builtin_prefetch(xpf, 0, 0);          // t+2 line toward L2
    float xn[4];
#pragma unroll
    for (int jj = 0; jj < 4; ++jj) xn[jj] = xnp[256 * jj];

    float cnt = 0.0f;
#pragma unroll
    for (int jj = 0; jj < 4; ++jj) {
      isyn[jj] = __builtin_fmaf(bsyn, isyn[jj], xv[jj]);
      v[jj]    = bmem * v[jj] + omb * isyn[jj];
      const bool fire = (v[jj] >= thv[jj]) && (refrac[jj] == 0);
      const float sj  = fire ? 1.0f : 0.0f;
      cnt += sj;
      ot[256 * jj] = sj;                    // spike output (B,T,N), fire-and-forget
      v[jj] -= sj * thv[jj];                // soft reset
      refrac[jj] = fire ? 2 : (refrac[jj] > 0 ? refrac[jj] - 1 : 0);
    }
    // Cluster fold across the 4 adjacent lanes sharing c (DPP, no LDS).
    cnt = quad_fold_add(cnt);
    if (h == 0) *aslot = (_Float16)(cnt * 0.0625f);   // cf[c], exact in f16
    barrier_lds_only();

    if (wq < 4) {
      const v16h a0 = *(const v16h*)&ldsA[0][ll][0];  // K = 0..31
      const v16h a1 = *(const v16h*)&ldsA[1][ll][0];  // K = 32..63
      v8f d = {};
      d = __builtin_amdgcn_wmma_f32_16x16x32_f16(false, a0, false, bf0,
                                                 (short)0, d, false, false);
      d = __builtin_amdgcn_wmma_f32_16x16x32_f16(false, a1, false, bf1,
                                                 (short)0, d, false, false);
      // D layout: M=0 lives in VGPR0 of lanes 0-15 (N = lane).
      if (ll < 16) ldsNS[wq * 16 + ll] = d[0];
    }
    barrier_lds_only();

    const float ns = ldsNS[c];              // lateral input for this cluster
#pragma unroll
    for (int jj = 0; jj < 4; ++jj) isyn[jj] += ns;
#pragma unroll
    for (int jj = 0; jj < 4; ++jj) xv[jj] = xn[jj];

    xt += N_NEUR; ot += N_NEUR;
  }
}

extern "C" void kernel_launch(void* const* d_in, const int* in_sizes, int n_in,
                              void* d_out, int out_size, void* d_ws, size_t ws_size,
                              hipStream_t stream) {
  (void)in_sizes; (void)n_in; (void)d_ws; (void)ws_size; (void)out_size;
  const float* x    = (const float*)d_in[0];
  const float* vth  = (const float*)d_in[1];
  const float* bm   = (const float*)d_in[2];
  const float* bs   = (const float*)d_in[3];
  const float* nw   = (const float*)d_in[4];
  const float* gain = (const float*)d_in[5];
  // d_in[6] (cluster_ids) is n % 64 by construction; hardcoded in the kernel.
  float* out = (float*)d_out;
  lif_assoc_kernel<<<dim3(64), dim3(256), 0, stream>>>(x, vth, bm, bs, nw, gain, out);
}
